// MiniRocketNative_9904194584958
// MI455X (gfx1250) — matrix operand order repaired and checked by reading, not compiled, not run on hardware
//
#include <hip/hip_runtime.h>

typedef __attribute__((ext_vector_type(16))) _Float16     v16h;
typedef __attribute__((ext_vector_type(8)))  float        v8f;
typedef __attribute__((ext_vector_type(8)))  unsigned int v8u;

#define SEQ_LEN 5000
#define C_IN    12
#define KLEN    9
#define KPD     1000
#define NDIL    10
#define BATCH   16
#define NTILES  313          // ceil(5000/16)
#define WTILES  157          // window 0: tiles [0,157), window 1: [157,313)
#define WAVES   8
#define THREADS 256
#define MBLK    32           // kernels per block = 2 WMMA M-tiles
#define GROUPS  32           // ceil(1000/32)

// K ordering (shared by A and B):
//  q = K/2 (pair), o = K&1.
//  q in [0,48):  chunk=q/16, p=q%16, hb=p/8, r=p%8 -> c = 4*chunk + 2*hb + r/4, j = 2*(r&3) + o
//  q in [48,64): p=q-48: o==0 && p<12 -> (c=p, j=8); else pad (A weight = 0)
__device__ __forceinline__ void decodeK(int K, int& c, int& j, bool& valid) {
    int q = K >> 1, o = K & 1;
    if (q < 48) {
        int chunk = q >> 4, p = q & 15, hb = (p >> 3) & 1, r = p & 7;
        c = 4 * chunk + 2 * hb + (r >> 2);
        j = 2 * (r & 3) + o;
        valid = true;
    } else {
        int p = q - 48;
        valid = (o == 0) && (p < 12);
        c = (p < 12) ? p : 0;
        j = 8;
    }
}

#define WMMA_F16(A, B, C) \
    __builtin_amdgcn_wmma_f32_16x16x32_f16(false, (A), false, (B), (short)0, (C), false, false)

// One 16x16 output tile for 2 M-tiles: 4 K-chunks on two independent accumulator
// chains (A: chunks 0,2; B: chunks 1,3) so no WMMA->WMMA same-dest hazard nops.
template<int D>
__device__ __forceinline__ void tile_mma(const unsigned int* pb0, const unsigned int* pb1,
                                         const unsigned int* pb2, const unsigned int* p30,
                                         const unsigned int* p31, const unsigned int* p32,
                                         const unsigned int* p33,
                                         const v16h (&af)[2][4],
                                         v8f& a0, v8f& a1, v8f& b0, v8f& b1)
{
    constexpr int YSR = WTILES * 16 + 8 * D + 16;
    constexpr int YS  = YSR + ((16 - (YSR & 31)) & 31);
    const v8f z = {};
    v8u bp;
    // chunk 0 -> chain A
    #pragma unroll
    for (int r = 0; r < 8; ++r)
        bp[r] = pb0[(r >> 2) * YS + 2 * D * (r & 3)];
    v16h bf = __builtin_bit_cast(v16h, bp);
    a0 = WMMA_F16(af[0][0], bf, z);
    a1 = WMMA_F16(af[1][0], bf, z);
    // chunk 1 -> chain B
    #pragma unroll
    for (int r = 0; r < 8; ++r)
        bp[r] = pb1[(r >> 2) * YS + 2 * D * (r & 3)];
    bf = __builtin_bit_cast(v16h, bp);
    b0 = WMMA_F16(af[0][1], bf, z);
    b1 = WMMA_F16(af[1][1], bf, z);
    // chunk 2 -> chain A
    #pragma unroll
    for (int r = 0; r < 8; ++r)
        bp[r] = pb2[(r >> 2) * YS + 2 * D * (r & 3)];
    bf = __builtin_bit_cast(v16h, bp);
    a0 = WMMA_F16(af[0][2], bf, a0);
    a1 = WMMA_F16(af[1][2], bf, a1);
    // chunk 3 (j=8 taps + pads; pad slots have zero A weights) -> chain B
    bp[0] = p30[0];  bp[1] = p30[YS];
    bp[2] = p31[0];  bp[3] = p31[YS];
    bp[4] = p32[0];  bp[5] = p32[YS];
    bp[6] = p33[0];  bp[7] = p33[YS];
    bf = __builtin_bit_cast(v16h, bp);
    b0 = WMMA_F16(af[0][3], bf, b0);
    b1 = WMMA_F16(af[1][3], bf, b1);
}

template<int D>
__global__ __launch_bounds__(THREADS)
void minirocket_wmma(const float* __restrict__ x,
                     const float* __restrict__ w,
                     const float* __restrict__ bias,
                     float* __restrict__ out,
                     int dil)
{
    // Window stride per channel (elements of packed f16x2); pad to YS%32==16 so the
    // two lane halves (base delta 2*YS) land in disjoint LDS bank groups.
    constexpr int YSR = WTILES * 16 + 8 * D + 16;
    constexpr int YS  = YSR + ((16 - (YSR & 31)) & 31);

    extern __shared__ char smem[];
    float*        red_max = (float*)smem;                      // [WAVES][MBLK]
    float*        red_cnt = red_max + WAVES * MBLK;            // [WAVES][MBLK]
    unsigned int* ys      = (unsigned int*)(smem + 2 * WAVES * MBLK * 4);

    const int tid  = threadIdx.x;
    const int wave = tid >> 5;
    const int ln   = tid & 31;
    const int n    = ln & 15;
    const int hi   = ln >> 4;

    const int grp = blockIdx.x;
    const int b   = blockIdx.y;
    const int k0g = grp * MBLK;

    // ---- A fragments: 2 M-tiles x 4 K-chunks (ternary weights exact in f16) ----
    v16h afrag[2][4];
    #pragma unroll
    for (int mt = 0; mt < 2; ++mt) {
        const int  k_lane = k0g + mt * 16 + n;
        const bool kv     = k_lane < KPD;
        #pragma unroll
        for (int kc = 0; kc < 4; ++kc) {
            v16h a = {};
            #pragma unroll
            for (int h = 0; h < 16; ++h) {
                const int r = h >> 1, odd = h & 1;
                const int lk = 16 * hi + 2 * r + odd;
                int c, j; bool valid;
                decodeK(kc * 32 + lk, c, j, valid);
                float wv = 0.0f;
                if (valid && kv)
                    wv = w[(((size_t)dil * KPD + k_lane) * C_IN + c) * KLEN + j];
                a[h] = (_Float16)wv;
            }
            afrag[mt][kc] = a;
        }
    }

    // ---- Per-lane bias rows (D-tile row M = mt*16 + hi*8 + r) ----
    float brow[2][8];
    #pragma unroll
    for (int mt = 0; mt < 2; ++mt)
        #pragma unroll
        for (int r = 0; r < 8; ++r) {
            int k = k0g + mt * 16 + hi * 8 + r;
            brow[mt][r] = (k < KPD) ? bias[dil * KPD + k] : 3.0e38f;
        }

    float rmax[2][8], rcnt[2][8];
    #pragma unroll
    for (int mt = 0; mt < 2; ++mt)
        #pragma unroll
        for (int r = 0; r < 8; ++r) { rmax[mt][r] = -3.4e38f; rcnt[mt][r] = 0.0f; }

    // chunk-3 base channels: hi=0 -> {0,2,4,6}; hi=1 -> {8,10, dummy(0), dummy(0)}
    const int c30 = hi ? 8  : 0;
    const int c31 = hi ? 10 : 2;
    const int c32 = hi ? 0  : 4;
    const int c33 = hi ? 0  : 6;

    for (int win = 0; win < 2; ++win) {
        const int ws = win * WTILES;
        const int we = (ws + WTILES < NTILES) ? (ws + WTILES) : NTILES;
        const int p0 = ws * 16;                 // window base in padded coords

        __syncthreads();                        // previous window fully consumed
        // ---- Stage packed pairs: ys[c][pp] = (xpad[p0+pp], xpad[p0+pp+D]) ----
        for (int idx = tid; idx < C_IN * YS; idx += THREADS) {
            const int c  = idx / YS;
            const int pp = idx - c * YS;
            const int t  = p0 + pp - 4 * D;
            const int t2 = t + D;
            float lo = (t  >= 0 && t  < SEQ_LEN) ? x[(b * C_IN + c) * SEQ_LEN + t ] : 0.0f;
            float hv = (t2 >= 0 && t2 < SEQ_LEN) ? x[(b * C_IN + c) * SEQ_LEN + t2] : 0.0f;
            union { _Float16 h[2]; unsigned int u; } pk;
            pk.h[0] = (_Float16)lo;
            pk.h[1] = (_Float16)hv;
            ys[idx] = pk.u;
        }
        __syncthreads();

        // ---- Per-wave base pointers (byte addresses in VGPRs, hi folded in) ----
        const int wloc = 16 * wave + n;         // window-local time of this lane
        const unsigned int* pb0 = ys + (0 + 2 * hi) * YS + wloc;   // chunks 0..2
        const unsigned int* pb1 = ys + (4 + 2 * hi) * YS + wloc;
        const unsigned int* pb2 = ys + (8 + 2 * hi) * YS + wloc;
        const unsigned int* p30 = ys + c30 * YS + wloc + 8 * D;    // chunk 3 (+8D folded)
        const unsigned int* p31 = ys + c31 * YS + wloc + 8 * D;
        const unsigned int* p32 = ys + c32 * YS + wloc + 8 * D;
        const unsigned int* p33 = ys + c33 * YS + wloc + 8 * D;

        // Full tiles only: no masking, no exec churn, EXEC all-1s at every WMMA.
        const int nfull = (we == NTILES) ? (NTILES - 1 - ws) : (we - ws);
        for (int it = wave; it < nfull; it += WAVES) {
            v8f a0, a1, b0, b1;
            tile_mma<D>(pb0, pb1, pb2, p30, p31, p32, p33, afrag, a0, a1, b0, b1);
            #pragma unroll
            for (int r = 0; r < 8; ++r) {
                const float v0 = a0[r] + b0[r];
                const float v1 = a1[r] + b1[r];
                rmax[0][r] = (v0 > rmax[0][r]) ? v0 : rmax[0][r];
                rmax[1][r] = (v1 > rmax[1][r]) ? v1 : rmax[1][r];
                rcnt[0][r] += (v0 > brow[0][r]) ? 1.0f : 0.0f;
                rcnt[1][r] += (v1 > brow[1][r]) ? 1.0f : 0.0f;
            }
            pb0 += 128; pb1 += 128; pb2 += 128;
            p30 += 128; p31 += 128; p32 += 128; p33 += 128;
        }

        // Tail tile 312 (t0 = 4992): handled once by the wave whose strided
        // pointers ended exactly there (nfull % WAVES); only n < 8 columns valid.
        if (we == NTILES && (nfull % WAVES) == wave) {
            v8f a0, a1, b0, b1;
            tile_mma<D>(pb0, pb1, pb2, p30, p31, p32, p33, afrag, a0, a1, b0, b1);
            if (n < 8) {
                #pragma unroll
                for (int r = 0; r < 8; ++r) {
                    const float v0 = a0[r] + b0[r];
                    const float v1 = a1[r] + b1[r];
                    rmax[0][r] = (v0 > rmax[0][r]) ? v0 : rmax[0][r];
                    rmax[1][r] = (v1 > rmax[1][r]) ? v1 : rmax[1][r];
                    rcnt[0][r] += (v0 > brow[0][r]) ? 1.0f : 0.0f;
                    rcnt[1][r] += (v1 > brow[1][r]) ? 1.0f : 0.0f;
                }
            }
        }
    }

    // ---- Reduce across the 16 time lanes of each half ----
    #pragma unroll
    for (int mt = 0; mt < 2; ++mt)
        #pragma unroll
        for (int r = 0; r < 8; ++r) {
            float m = rmax[mt][r], cn = rcnt[mt][r];
            #pragma unroll
            for (int mask = 1; mask < 16; mask <<= 1) {
                const float ms = __shfl_xor(m, mask, 32);
                m  = (ms > m) ? ms : m;
                cn += __shfl_xor(cn, mask, 32);
            }
            if (n == 0) {
                red_max[wave * MBLK + mt * 16 + hi * 8 + r] = m;
                red_cnt[wave * MBLK + mt * 16 + hi * 8 + r] = cn;
            }
        }
    __syncthreads();

    // ---- Cross-wave reduce + write [maxv(1000) | ppv(1000)] per dilation ----
    if (tid < MBLK) {
        float m = -3.4e38f, cn = 0.0f;
        #pragma unroll
        for (int w2 = 0; w2 < WAVES; ++w2) {
            const float ms = red_max[w2 * MBLK + tid];
            m  = (ms > m) ? ms : m;
            cn += red_cnt[w2 * MBLK + tid];
        }
        const int k = k0g + tid;
        if (k < KPD) {
            float* ob = out + (size_t)b * (2 * NDIL * KPD) + (size_t)dil * (2 * KPD);
            ob[k]       = m;
            ob[KPD + k] = cn * (1.0f / SEQ_LEN);
        }
    }
}

extern "C" void kernel_launch(void* const* d_in, const int* in_sizes, int n_in,
                              void* d_out, int out_size, void* d_ws, size_t ws_size,
                              hipStream_t stream) {
    (void)in_sizes; (void)n_in; (void)d_ws; (void)ws_size; (void)out_size;
    const float* x    = (const float*)d_in[0];   // [16,12,5000]
    const float* w    = (const float*)d_in[1];   // [10,1000,12,9]
    const float* bias = (const float*)d_in[2];   // [10,1000]
    float*       out  = (float*)d_out;           // [16,20000]

    dim3 grid(GROUPS, BATCH), block(THREADS);

#define LAUNCH_DIL(IDX, DD)                                                          \
    do {                                                                             \
        constexpr int ysr = WTILES * 16 + 8 * (DD) + 16;                             \
        constexpr int ysz = ysr + ((16 - (ysr & 31)) & 31);                          \
        size_t shmem = 2 * WAVES * MBLK * 4 + (size_t)C_IN * ysz * 4;                \
        (void)hipFuncSetAttribute((const void*)&minirocket_wmma<DD>,                 \
                                  hipFuncAttributeMaxDynamicSharedMemorySize,        \
                                  (int)shmem);                                       \
        minirocket_wmma<DD><<<grid, block, shmem, stream>>>(x, w, bias, out, IDX);   \
    } while (0)

    LAUNCH_DIL(0, 1);
    LAUNCH_DIL(1, 2);
    LAUNCH_DIL(2, 4);
    LAUNCH_DIL(3, 8);
    LAUNCH_DIL(4, 16);
    LAUNCH_DIL(5, 32);
    LAUNCH_DIL(6, 64);
    LAUNCH_DIL(7, 128);
    LAUNCH_DIL(8, 256);
    LAUNCH_DIL(9, 512);
#undef LAUNCH_DIL
}
